// Attention_29729763623354
// MI455X (gfx1250) — matrix operand order repaired and checked by reading, not compiled, not run on hardware
//
#include <hip/hip_runtime.h>

// Problem constants (match reference)
#define BB 2
#define NN 2048
#define MM 2048
#define DD 1024
#define HH 16
#define HD 64

typedef _Float16 v8h  __attribute__((ext_vector_type(8)));
typedef _Float16 v16h __attribute__((ext_vector_type(16)));
typedef float    v8f  __attribute__((ext_vector_type(8)));

union V16H { v16h v; v8h h[2]; };

static __device__ __forceinline__ v8f wmma_f16(v16h a, v16h b, v8f c) {
  // D = A(16x32 f16) * B(32x16 f16) + C(16x16 f32)
  return __builtin_amdgcn_wmma_f32_16x16x32_f16(false, a, false, b, (short)0, c,
                                                false, false);
}

// ---------------------------------------------------------------------------
// CDNA5 async global->LDS copy (ASYNCcnt-tracked), with safe fallback.
// ---------------------------------------------------------------------------
typedef int v4i __attribute__((vector_size(4 * sizeof(int))));
typedef __attribute__((address_space(1))) v4i* gv4i_p;
typedef __attribute__((address_space(3))) v4i* lv4i_p;

static __device__ __forceinline__ void async_copy_b128(const _Float16* g,
                                                       _Float16* l) {
#if __has_builtin(__builtin_amdgcn_global_load_async_to_lds_b128)
  __builtin_amdgcn_global_load_async_to_lds_b128(
      (gv4i_p)(void*)(_Float16*)g, (lv4i_p)(void*)l, 0, 0);
#else
  *(v8h*)l = *(const v8h*)g;   // sync fallback: global_load + ds_store
#endif
}

static __device__ __forceinline__ void wait_asynccnt_le2() {
#if __has_builtin(__builtin_amdgcn_s_wait_asynccnt)
  __builtin_amdgcn_s_wait_asynccnt(2);
#elif __has_builtin(__builtin_amdgcn_global_load_async_to_lds_b128)
  asm volatile("s_wait_asynccnt 0x2" ::: "memory");
#endif
}

// ---------------------------------------------------------------------------
// Elementwise f32 -> f16 conversion
// ---------------------------------------------------------------------------
__global__ __launch_bounds__(256) void f32_to_f16_kernel(
    const float* __restrict__ in, _Float16* __restrict__ out, int n) {
  int i = blockIdx.x * 256 + threadIdx.x;
  if (i < n) out[i] = (_Float16)in[i];
}

// ---------------------------------------------------------------------------
// Register-blocked GEMM: out = (X @ W^T + bias) * scale
//   X: [B*seq, D] f16 row-major, W: [D, D] f16 row-major (W[dout][k])
// Block = 4 waves, tile 128(M) x 64(N). Each wave: 32x64 -> 8 named f32
// accumulators (kept in fixed VGPRs to avoid WMMA D-copy hazards),
// 8 wmma per 32-K step. Weight tile (64 rows x 32 K = 4 KB) is shared by all
// waves and double-buffered in LDS via async global->LDS loads.
//   mode 0: f16 head-split out[b][h][n][hd]        (Q, K)
//   mode 1: f16 transposed out[b][h][hd][m]        (V)
//   mode 2: f32 row-major out32[m][dout] (+bias)   (output projection)
// ---------------------------------------------------------------------------
__global__ __launch_bounds__(128) void gemm_kernel(
    const _Float16* __restrict__ X, const _Float16* __restrict__ W,
    const float* __restrict__ bias, _Float16* __restrict__ out16,
    float* __restrict__ out32, float scale, int mode) {
  __shared__ _Float16 bt[2][64 * 32];   // [buf][row][k] : 4 KB per buffer

  const int tid  = threadIdx.x;
  const int wid  = tid >> 5;
  const int lane = tid & 31;
  const int grp  = lane >> 4;
  const int l16  = lane & 15;
  const int m0 = blockIdx.x * 128 + wid * 32;   // this wave's 32 rows
  const int n0 = blockIdx.y * 64;               // block's 64 output cols

  const _Float16* Wt = W + (size_t)n0 * DD;     // 64 weight rows for this block

  // Stage K-step 0 into buffer 0: 64 rows x 64 B = 256 x 16 B chunks,
  // 2 chunks per thread -> 2 async instructions per wave.
  {
    const int c0 = tid, c1 = tid + 128;
    async_copy_b128(Wt + (size_t)(c0 >> 2) * DD + (c0 & 3) * 8,
                    &bt[0][(c0 >> 2) * 32 + (c0 & 3) * 8]);
    async_copy_b128(Wt + (size_t)(c1 >> 2) * DD + (c1 & 3) * 8,
                    &bt[0][(c1 >> 2) * 32 + (c1 & 3) * 8]);
  }

  const _Float16* arow0 = X + (size_t)(m0 + l16) * DD;
  const _Float16* arow1 = X + (size_t)(m0 + 16 + l16) * DD;

  // Named accumulators: mt x nt = 2 x 4
  v8f c00 = {}, c01 = {}, c02 = {}, c03 = {};
  v8f c10 = {}, c11 = {}, c12 = {}, c13 = {};

  for (int ks = 0; ks < DD / 32; ++ks) {
    const int k = ks * 32;
    const int cur = ks & 1;

    // Prefetch next weight tile into the other buffer (async).
    if (ks + 1 < DD / 32) {
      const _Float16* Wn = Wt + (size_t)(k + 32);
      _Float16* lb = &bt[cur ^ 1][0];
      const int c0 = tid, c1 = tid + 128;
      async_copy_b128(Wn + (size_t)(c0 >> 2) * DD + (c0 & 3) * 8,
                      &lb[(c0 >> 2) * 32 + (c0 & 3) * 8]);
      async_copy_b128(Wn + (size_t)(c1 >> 2) * DD + (c1 & 3) * 8,
                      &lb[(c1 >> 2) * 32 + (c1 & 3) * 8]);
    }

    // Wait for the current buffer's 2 async ops (the 2 newest may still fly).
    wait_asynccnt_le2();
    __syncthreads();

    // A fragments for this wave's two 16-row tiles.
    V16H a0, a1;
    a0.h[0] = *(const v8h*)(arow0 + k + grp * 16);
    a0.h[1] = *(const v8h*)(arow0 + k + 8 + grp * 16);
    a1.h[0] = *(const v8h*)(arow1 + k + grp * 16);
    a1.h[1] = *(const v8h*)(arow1 + k + 8 + grp * 16);

    const _Float16* lbase = &bt[cur][(size_t)l16 * 32 + grp * 16];
    v16h b0 = *(const v16h*)(lbase + 0 * 16 * 32);
    v16h b1 = *(const v16h*)(lbase + 1 * 16 * 32);
    v16h b2 = *(const v16h*)(lbase + 2 * 16 * 32);
    v16h b3 = *(const v16h*)(lbase + 3 * 16 * 32);

    c00 = wmma_f16(a0.v, b0, c00);
    c10 = wmma_f16(a1.v, b0, c10);
    c01 = wmma_f16(a0.v, b1, c01);
    c11 = wmma_f16(a1.v, b1, c11);
    c02 = wmma_f16(a0.v, b2, c02);
    c12 = wmma_f16(a1.v, b2, c12);
    c03 = wmma_f16(a0.v, b3, c03);
    c13 = wmma_f16(a1.v, b3, c13);

    __syncthreads();   // all waves done with bt[cur] before it is restaged
  }

  // Epilogue (runs once; copies here are cheap)
  const v8f accs[2][4] = {{c00, c01, c02, c03}, {c10, c11, c12, c13}};
#pragma unroll
  for (int mt = 0; mt < 2; ++mt) {
#pragma unroll
    for (int nt = 0; nt < 4; ++nt) {
      const int dout = n0 + nt * 16 + l16;
      const float bv = bias[dout];
      const int h  = dout / HD;
      const int hd = dout % HD;
#pragma unroll
      for (int r = 0; r < 8; ++r) {
        const int mg = m0 + mt * 16 + grp * 8 + r;
        const float val = (accs[mt][nt][r] + bv) * scale;
        if (mode == 2) {
          out32[(size_t)mg * DD + dout] = val;
        } else {
          const int b = mg / NN;
          const int n = mg % NN;
          const size_t idx = (mode == 1)
              ? (((size_t)b * HH + h) * HD + hd) * MM + n
              : (((size_t)b * HH + h) * NN + n) * HD + hd;
          out16[idx] = (_Float16)val;
        }
      }
    }
  }
}

// ---------------------------------------------------------------------------
// Flash attention: one wave per (b, h, 16 query rows).
//   Qh: [B*H][N][HD]  Kh: [B*H][M][HD]  Vt: [B*H][HD][M]
//   Oc: [B][N][D] f16 (merged heads), softmax-normalized.
// Per 32-key step: 4 wmma for S (16x32), online softmax, P via LDS repack,
// 4 wmma for O += P*V (16x64).
// ---------------------------------------------------------------------------
__global__ __launch_bounds__(32) void attn_kernel(
    const _Float16* __restrict__ Qh, const _Float16* __restrict__ Kh,
    const _Float16* __restrict__ Vt, _Float16* __restrict__ Oc) {
  __shared__ _Float16 pbuf[16 * 32];
  const int lane = threadIdx.x & 31;
  const int grp  = lane >> 4;
  const int l16  = lane & 15;
  const int q0 = blockIdx.x * 16;
  const int h  = blockIdx.y;
  const int b  = blockIdx.z;
  const size_t bh = (size_t)b * HH + h;

  // Q A-fragments: hd 0..31 and hd 32..63 (scale folded in projection)
  V16H aq0, aq1;
  {
    const _Float16* qrow = Qh + (bh * NN + q0 + l16) * HD;
    aq0.h[0] = *(const v8h*)(qrow + grp * 16);
    aq0.h[1] = *(const v8h*)(qrow + 8 + grp * 16);
    aq1.h[0] = *(const v8h*)(qrow + 32 + grp * 16);
    aq1.h[1] = *(const v8h*)(qrow + 32 + 8 + grp * 16);
  }

  v8f o0 = {}, o1 = {}, o2 = {}, o3 = {};
  float mrow[8], lrow[8];
#pragma unroll
  for (int r = 0; r < 8; ++r) { mrow[r] = -3.0e38f; lrow[r] = 0.0f; }

  const _Float16* kbase = Kh + bh * (size_t)MM * HD;
  const _Float16* vbase = Vt + bh * (size_t)HD * MM;

  for (int kb = 0; kb < MM; kb += 32) {
    // Prefetch next K tile (speculative, dropped if OOB)
    {
      int pk = (kb + 32 < MM) ? (kb + 32) : kb;
      __builtin_prefetch(kbase + (size_t)(pk + l16) * HD, 0, 3);
    }
    // S = Q * K^T : two 16x16 tiles (keys kb..+15 and kb+16..+31)
    v8f s0 = {}, s1 = {};
    {
      v16h bk;
      bk = *(const v16h*)(kbase + (size_t)(kb + l16) * HD + grp * 16);
      s0 = wmma_f16(aq0.v, bk, s0);
      bk = *(const v16h*)(kbase + (size_t)(kb + l16) * HD + 32 + grp * 16);
      s0 = wmma_f16(aq1.v, bk, s0);
      bk = *(const v16h*)(kbase + (size_t)(kb + 16 + l16) * HD + grp * 16);
      s1 = wmma_f16(aq0.v, bk, s1);
      bk = *(const v16h*)(kbase + (size_t)(kb + 16 + l16) * HD + 32 + grp * 16);
      s1 = wmma_f16(aq1.v, bk, s1);
    }

    // Online softmax: row r of C lives in one 16-lane half
    float alpha[8];
    v8f p0, p1;
#pragma unroll
    for (int r = 0; r < 8; ++r) {
      float mx = fmaxf(s0[r], s1[r]);
      mx = fmaxf(mx, __shfl_xor(mx, 8, 32));
      mx = fmaxf(mx, __shfl_xor(mx, 4, 32));
      mx = fmaxf(mx, __shfl_xor(mx, 2, 32));
      mx = fmaxf(mx, __shfl_xor(mx, 1, 32));
      float mnew = fmaxf(mrow[r], mx);
      alpha[r] = __expf(mrow[r] - mnew);
      p0[r] = __expf(s0[r] - mnew);
      p1[r] = __expf(s1[r] - mnew);
      float rs = p0[r] + p1[r];
      rs += __shfl_xor(rs, 8, 32);
      rs += __shfl_xor(rs, 4, 32);
      rs += __shfl_xor(rs, 2, 32);
      rs += __shfl_xor(rs, 1, 32);
      lrow[r] = lrow[r] * alpha[r] + rs;
      mrow[r] = mnew;
    }
#pragma unroll
    for (int r = 0; r < 8; ++r) {
      o0[r] *= alpha[r]; o1[r] *= alpha[r];
      o2[r] *= alpha[r]; o3[r] *= alpha[r];
    }

    // Repack P (16 rows x 32 keys) through LDS into an A-fragment
#pragma unroll
    for (int r = 0; r < 8; ++r) {
      int row = grp * 8 + r;
      pbuf[row * 32 + l16]      = (_Float16)p0[r];
      pbuf[row * 32 + 16 + l16] = (_Float16)p1[r];
    }
    __syncthreads();
    V16H ap;
    ap.h[0] = *(const v8h*)(&pbuf[l16 * 32 + grp * 16]);
    ap.h[1] = *(const v8h*)(&pbuf[l16 * 32 + 8 + grp * 16]);

    // O(16x64) += P(16x32) * V(32x64) : 4 wmma (Vt makes B-frags contiguous)
    v16h bv;
    bv = *(const v16h*)(vbase + (size_t)(l16) * MM + kb + grp * 16);
    o0 = wmma_f16(ap.v, bv, o0);
    bv = *(const v16h*)(vbase + (size_t)(16 + l16) * MM + kb + grp * 16);
    o1 = wmma_f16(ap.v, bv, o1);
    bv = *(const v16h*)(vbase + (size_t)(32 + l16) * MM + kb + grp * 16);
    o2 = wmma_f16(ap.v, bv, o2);
    bv = *(const v16h*)(vbase + (size_t)(48 + l16) * MM + kb + grp * 16);
    o3 = wmma_f16(ap.v, bv, o3);
    __syncthreads();
  }

  // Normalize and store merged-head output [B, N, D] as f16
#pragma unroll
  for (int r = 0; r < 8; ++r) {
    float il = 1.0f / lrow[r];
    size_t base = ((size_t)b * NN + q0 + grp * 8 + r) * DD + (size_t)h * HD;
    Oc[base + l16]      = (_Float16)(o0[r] * il);
    Oc[base + 16 + l16] = (_Float16)(o1[r] * il);
    Oc[base + 32 + l16] = (_Float16)(o2[r] * il);
    Oc[base + 48 + l16] = (_Float16)(o3[r] * il);
  }
}

// ---------------------------------------------------------------------------
extern "C" void kernel_launch(void* const* d_in, const int* in_sizes, int n_in,
                              void* d_out, int out_size, void* d_ws, size_t ws_size,
                              hipStream_t stream) {
  const float* q_in = (const float*)d_in[0];
  const float* k_in = (const float*)d_in[1];
  const float* v_in = (const float*)d_in[2];
  const float* Wq   = (const float*)d_in[3];
  const float* bq   = (const float*)d_in[4];
  const float* Wk   = (const float*)d_in[5];
  const float* bk   = (const float*)d_in[6];
  const float* Wv   = (const float*)d_in[7];
  const float* bv   = (const float*)d_in[8];
  const float* Wp   = (const float*)d_in[9];
  const float* bp   = (const float*)d_in[10];
  float* out = (float*)d_out;

  // Workspace carve-up (f16 elements): 3 activations + 4 weights + Q/K/V/attn
  _Float16* ws = (_Float16*)d_ws;
  const size_t SZ_X = (size_t)BB * NN * DD;  // 4 Mi elems each
  const size_t SZ_W = (size_t)DD * DD;       // 1 Mi elems each
  _Float16* q16  = ws;
  _Float16* k16  = q16 + SZ_X;
  _Float16* v16  = k16 + SZ_X;
  _Float16* wq16 = v16 + SZ_X;
  _Float16* wk16 = wq16 + SZ_W;
  _Float16* wv16 = wk16 + SZ_W;
  _Float16* wp16 = wv16 + SZ_W;
  _Float16* Qh   = wp16 + SZ_W;  // [B*H][N][HD]
  _Float16* Kh   = Qh + SZ_X;    // [B*H][M][HD]
  _Float16* Vt   = Kh + SZ_X;    // [B*H][HD][M]
  _Float16* A16  = Vt + SZ_X;    // [B][N][D] attention output (pre-proj)

  // 1) convert inputs + weights to f16
  f32_to_f16_kernel<<<(int)((SZ_X + 255) / 256), 256, 0, stream>>>(q_in, q16, (int)SZ_X);
  f32_to_f16_kernel<<<(int)((SZ_X + 255) / 256), 256, 0, stream>>>(k_in, k16, (int)SZ_X);
  f32_to_f16_kernel<<<(int)((SZ_X + 255) / 256), 256, 0, stream>>>(v_in, v16, (int)SZ_X);
  f32_to_f16_kernel<<<(int)((SZ_W + 255) / 256), 256, 0, stream>>>(Wq, wq16, (int)SZ_W);
  f32_to_f16_kernel<<<(int)((SZ_W + 255) / 256), 256, 0, stream>>>(Wk, wk16, (int)SZ_W);
  f32_to_f16_kernel<<<(int)((SZ_W + 255) / 256), 256, 0, stream>>>(Wv, wv16, (int)SZ_W);
  f32_to_f16_kernel<<<(int)((SZ_W + 255) / 256), 256, 0, stream>>>(Wp, wp16, (int)SZ_W);

  // 2) projections (scale = HD^-0.5 folded into Q; V stored transposed)
  dim3 gproj(BB * NN / 128, DD / 64);
  dim3 blk(128);
  gemm_kernel<<<gproj, blk, 0, stream>>>(q16, wq16, bq, Qh, nullptr, 0.125f, 0);
  gemm_kernel<<<gproj, blk, 0, stream>>>(k16, wk16, bk, Kh, nullptr, 1.0f, 0);
  gemm_kernel<<<gproj, blk, 0, stream>>>(v16, wv16, bv, Vt, nullptr, 1.0f, 1);

  // 3) flash attention
  dim3 gattn(NN / 16, HH, BB);
  attn_kernel<<<gattn, dim3(32), 0, stream>>>(Qh, Kh, Vt, A16);

  // 4) output projection -> f32
  gemm_kernel<<<gproj, blk, 0, stream>>>(A16, wp16, bp, nullptr, out, 1.0f, 2);
}
